// LSTM_76751065579883
// MI455X (gfx1250) — matrix operand order repaired and checked by reading, not compiled, not run on hardware
//
#include <hip/hip_runtime.h>

// ---------------------------------------------------------------------------
// LSTM forward for MI455X (gfx1250): bf16 WMMA (v_wmma_f32_16x16x32_bf16),
// persistent recurrent kernel, Wh + h staged in LDS via async global->LDS
// DMA (ASYNCcnt path), atomic grid barrier, global_prefetch for xg gather.
// ---------------------------------------------------------------------------

#define BATCH 32
#define SEQ   512
#define ISZ   512
#define HID   1024
#define G4    4096          // 4*HID
#define NWG   64            // persistent workgroups in recurrence (HID/16)
#define LDSK  1032          // 1024 + 8 bf16 pad -> bank-conflict-free b128 LDS reads

typedef __attribute__((ext_vector_type(16))) __bf16 v16bf;
typedef __attribute__((ext_vector_type(8)))  __bf16 v8bf;
typedef __attribute__((ext_vector_type(8)))  float  v8f;

union ABf16 { v16bf v; v8bf h[2]; };

// ---- workspace layout (bytes) ---------------------------------------------
#define BAR_OFF   ((size_t)0)
#define HBUF_OFF  ((size_t)512)                              // 2*32*1024 bf16
#define XB_OFF    (HBUF_OFF + (size_t)2*BATCH*HID*2)         // x bf16: 16 MB
#define WXB_OFF   (XB_OFF + (size_t)BATCH*SEQ*ISZ*2)         // Wx bf16: 4 MB
#define XG_OFF    (WXB_OFF + (size_t)G4*ISZ*2)               // xg f32: 256 MB
#define WHB_OFF   (XG_OFF + (size_t)BATCH*SEQ*G4*4)          // Wh bf16: 8 MB

__device__ __forceinline__ unsigned short f2bf(float f) {
    unsigned u = __float_as_uint(f);
    u = (u + 0x7FFFu + ((u >> 16) & 1u)) >> 16;   // round-to-nearest-even
    return (unsigned short)u;
}
__device__ __forceinline__ float sigm_f(float x) {
    return 1.0f / (1.0f + __expf(-x));
}
__device__ __forceinline__ float tanh_f(float x) {
    float e = __expf(2.0f * x);
    return 1.0f - 2.0f / (e + 1.0f);
}

// Async 16-byte Global -> LDS copy (CDNA5; tracked by ASYNCcnt).
__device__ __forceinline__ void async_g2l_b128(unsigned ldsOff, const void* gptr) {
    asm volatile("global_load_async_to_lds_b128 %0, %1, off"
                 :: "v"(ldsOff), "v"((unsigned long long)(size_t)gptr)
                 : "memory");
}
__device__ __forceinline__ void wait_async0() {
    asm volatile("s_wait_asynccnt 0x0" ::: "memory");
}

// ---- f32 -> bf16 conversion -----------------------------------------------
__global__ void cvt_bf16_kernel(const float* __restrict__ in,
                                unsigned short* __restrict__ out, int n) {
    int i = blockIdx.x * blockDim.x + threadIdx.x;
    if (i < n) out[i] = f2bf(in[i]);
}

// ---- zero h ping-pong buffers + barrier words (fresh every launch) --------
__global__ void lstm_init_kernel(unsigned short* __restrict__ hbuf,
                                 unsigned* __restrict__ bar) {
    int i = blockIdx.x * blockDim.x + threadIdx.x;
    if (i < 2 * BATCH * HID) hbuf[i] = 0;
    if (i < 2) bar[i] = 0u;
}

// ---- Phase 1: xg[m][n] = sum_k x[m][k]*Wx[n][k] + bx[n] + bh[n] -----------
// m = b*SEQ + t (16384 rows), n in [0,4096), K = 512.
// Block: 256 thr = 8 waves; wave computes a 16x64 strip (4 WMMA tiles).
__global__ void lstm_xgemm_kernel(const unsigned short* __restrict__ xb,
                                  const unsigned short* __restrict__ wxb,
                                  const float* __restrict__ bx,
                                  const float* __restrict__ bh,
                                  float* __restrict__ xg) {
    const int wave = threadIdx.x >> 5;
    const int lane = threadIdx.x & 31;
    const int lo16 = (lane < 16) ? 1 : 0;
    const int m0 = blockIdx.y * 128 + wave * 16;
    const int n0 = blockIdx.x * 64;

    v8f acc[4] = {};
    const int kbA = lo16 ? 0 : 8;    // A-frag K sub-offsets (16-bit layout)
    const int kbB = lo16 ? 0 : 8;    // B-frag K sub-offsets

    const unsigned short* arow = xb + (size_t)(m0 + (lane & 15)) * ISZ;

    for (int kc = 0; kc < ISZ / 32; ++kc) {
        ABf16 a;
        a.h[0] = *(const v8bf*)(arow + kc * 32 + kbA);
        a.h[1] = *(const v8bf*)(arow + kc * 32 + kbA + 16);
#pragma unroll
        for (int nt = 0; nt < 4; ++nt) {
            const unsigned short* brow =
                wxb + (size_t)(n0 + nt * 16 + (lane & 15)) * ISZ;
            ABf16 b;
            b.h[0] = *(const v8bf*)(brow + kc * 32 + kbB);
            b.h[1] = *(const v8bf*)(brow + kc * 32 + kbB + 16);
            acc[nt] = __builtin_amdgcn_wmma_f32_16x16x32_bf16(
                false, a.v, false, b.v, (short)0, acc[nt], false, false);
        }
    }
#pragma unroll
    for (int nt = 0; nt < 4; ++nt) {
        const int n = n0 + nt * 16 + (lane & 15);
        const float bias = bx[n] + bh[n];
#pragma unroll
        for (int e = 0; e < 8; ++e) {
            const int m = m0 + e + (lo16 ? 0 : 8);
            xg[(size_t)m * G4 + n] = acc[nt][e] + bias;
        }
    }
}

// ---- Phase 2: persistent recurrence ---------------------------------------
// 64 WGs x 256 thr. WG jtile owns hidden columns [jtile*16, jtile*16+16) of
// all 4 gates; its bf16 Wh slice (4*16 rows x 1024 K) is async-DMA'd into
// LDS once; h(t) is async-DMA'd into LDS each step. Wave w = (mtile = w&1,
// gate = w>>1); inner K loop is pure LDS b128 reads -> WMMA.
__global__ void lstm_recurrent_kernel(const unsigned short* __restrict__ whb,
                                      const float* __restrict__ xg,
                                      unsigned short* __restrict__ hbuf,
                                      float* __restrict__ out,
                                      unsigned* __restrict__ bar) {
    __shared__ __align__(16) unsigned short whLds[64 * LDSK];
    __shared__ __align__(16) unsigned short hLds[BATCH * LDSK];
    __shared__ float gLds[4][BATCH][16];

    const int jtile = blockIdx.x;
    const int wave  = threadIdx.x >> 5;
    const int lane  = threadIdx.x & 31;
    const int mtile = wave & 1;
    const int gate  = wave >> 1;
    const int lo16  = (lane < 16) ? 1 : 0;

    const unsigned whBase = (unsigned)(unsigned long long)(const void*)whLds;
    const unsigned hBase  = (unsigned)(unsigned long long)(const void*)hLds;

    // One-time async DMA of this WG's Wh slice into LDS (row r = gate*16+c),
    // 16 bytes per transfer so the +8 bf16 row skew is preserved.
    for (int i = threadIdx.x; i < 64 * 128; i += 256) {      // 8192 x 16B
        const int r  = i >> 7;                               // 0..63
        const int kk = (i & 127) << 3;                       // bf16 index
        const int g = r >> 4, c = r & 15;
        const unsigned short* src =
            whb + (size_t)(g * HID + jtile * 16 + c) * HID + kk;
        async_g2l_b128(whBase + (unsigned)(r * LDSK + kk) * 2u, src);
    }
    wait_async0();
    __syncthreads();

    // Per-thread cell state: 2 of the 32x16 (b,col) cells this WG owns.
    const int e0 = threadIdx.x, e1 = threadIdx.x + 256;
    const int b0 = e0 >> 4, q0 = e0 & 15;
    const int b1 = e1 >> 4, q1 = e1 & 15;
    float cst0 = 0.0f, cst1 = 0.0f;

    const int kbA = lo16 ? 0 : 8;
    const int kbB = lo16 ? 0 : 8;
    const unsigned short* arow = hLds + (size_t)(mtile * 16 + (lane & 15)) * LDSK;
    const unsigned short* brow = whLds + (size_t)(gate * 16 + (lane & 15)) * LDSK;
    const int colw = jtile * 16 + (lane & 15);        // this lane's gate column
    const int gcol = gate * HID + colw;               // column in [0,4096)

    for (int t = 0; t < SEQ; ++t) {
        const unsigned short* hsrc = hbuf + (size_t)(t & 1) * BATCH * HID;

        // Stage h(t) [32x1024 bf16] into LDS via async DMA (read once per WG,
        // shared by all 8 waves instead of 4x redundant global reads).
        for (int i = threadIdx.x; i < BATCH * 128; i += 256) {   // 4096 x 16B
            const int r  = i >> 7;                               // batch row
            const int kk = (i & 127) << 3;
            async_g2l_b128(hBase + (unsigned)(r * LDSK + kk) * 2u,
                           hsrc + (size_t)r * HID + kk);
        }

        // Prefetch this wave's xg gather lines for step t (consumed after
        // the K loop; stride between rows is 8 MB -> 8 distinct lines).
#pragma unroll
        for (int e = 0; e < 8; ++e) {
            const int bm = mtile * 16 + e + (lo16 ? 0 : 8);
            __builtin_prefetch(&xg[((size_t)bm * SEQ + t) * G4 + gcol], 0, 1);
        }

        wait_async0();
        __syncthreads();

        v8f acc = {};
#pragma unroll 4
        for (int kc = 0; kc < HID / 32; ++kc) {
            ABf16 a, b;
            a.h[0] = *(const v8bf*)(arow + kc * 32 + kbA);
            a.h[1] = *(const v8bf*)(arow + kc * 32 + kbA + 16);
            b.h[0] = *(const v8bf*)(brow + kc * 32 + kbB);
            b.h[1] = *(const v8bf*)(brow + kc * 32 + kbB + 16);
            acc = __builtin_amdgcn_wmma_f32_16x16x32_bf16(
                false, a.v, false, b.v, (short)0, acc, false, false);
        }

        // Add precomputed input projection, apply gate nonlinearity, share.
#pragma unroll
        for (int e = 0; e < 8; ++e) {
            const int bm = mtile * 16 + e + (lo16 ? 0 : 8);  // batch row
            float v = acc[e] + xg[((size_t)bm * SEQ + t) * G4 + gcol];
            v = (gate == 2) ? tanh_f(v) : sigm_f(v);
            gLds[gate][bm][lane & 15] = v;
        }
        __syncthreads();

        // c/h update: each thread owns 2 fixed (b,col) cells.
        {
            const float iv = gLds[0][b0][q0], fv = gLds[1][b0][q0];
            const float gv = gLds[2][b0][q0], ov = gLds[3][b0][q0];
            cst0 = cst0 * fv + iv * gv;
            const float hv = ov * tanh_f(cst0);
            out[((size_t)b0 * SEQ + t) * HID + jtile * 16 + q0] = hv;
            hbuf[(size_t)((t + 1) & 1) * BATCH * HID + b0 * HID + jtile * 16 + q0] = f2bf(hv);
        }
        {
            const float iv = gLds[0][b1][q1], fv = gLds[1][b1][q1];
            const float gv = gLds[2][b1][q1], ov = gLds[3][b1][q1];
            cst1 = cst1 * fv + iv * gv;
            const float hv = ov * tanh_f(cst1);
            out[((size_t)b1 * SEQ + t) * HID + jtile * 16 + q1] = hv;
            hbuf[(size_t)((t + 1) & 1) * BATCH * HID + b1 * HID + jtile * 16 + q1] = f2bf(hv);
        }

        // Grid-wide barrier: monotonic arrival counter + generation word.
        __syncthreads();
        __threadfence();
        if (threadIdx.x == 0) {
            const unsigned want = (unsigned)(t + 1);
            const unsigned prev = atomicAdd(bar, 1u);
            if (prev == (unsigned)NWG * want - 1u) {
                __atomic_store_n(bar + 1, want, __ATOMIC_RELEASE);
            } else {
                while (__atomic_load_n(bar + 1, __ATOMIC_ACQUIRE) < want) {
                    __builtin_amdgcn_s_sleep(2);
                }
            }
        }
        __syncthreads();
    }
}

// ---------------------------------------------------------------------------
extern "C" void kernel_launch(void* const* d_in, const int* in_sizes, int n_in,
                              void* d_out, int out_size, void* d_ws, size_t ws_size,
                              hipStream_t stream) {
    const float* x  = (const float*)d_in[0];   // [32,512,512]
    const float* Wx = (const float*)d_in[1];   // [4096,512]
    const float* bx = (const float*)d_in[2];   // [4096]
    const float* Wh = (const float*)d_in[3];   // [4096,1024]
    const float* bh = (const float*)d_in[4];   // [4096]
    float* out = (float*)d_out;                // [32,512,1024]

    char* ws = (char*)d_ws;
    unsigned*       bar  = (unsigned*)(ws + BAR_OFF);
    unsigned short* hbuf = (unsigned short*)(ws + HBUF_OFF);
    unsigned short* xb   = (unsigned short*)(ws + XB_OFF);
    unsigned short* wxb  = (unsigned short*)(ws + WXB_OFF);
    float*          xg   = (float*)(ws + XG_OFF);
    unsigned short* whb  = (unsigned short*)(ws + WHB_OFF);

    const int NX  = BATCH * SEQ * ISZ;   // 8388608
    const int NWX = G4 * ISZ;            // 2097152
    const int NWH = G4 * HID;            // 4194304

    cvt_bf16_kernel<<<(NX + 255) / 256, 256, 0, stream>>>(x, xb, NX);
    cvt_bf16_kernel<<<(NWX + 255) / 256, 256, 0, stream>>>(Wx, wxb, NWX);
    cvt_bf16_kernel<<<(NWH + 255) / 256, 256, 0, stream>>>(Wh, whb, NWH);
    lstm_init_kernel<<<256, 256, 0, stream>>>(hbuf, bar);

    lstm_xgemm_kernel<<<dim3(G4 / 64, (BATCH * SEQ) / 128), 256, 0, stream>>>(
        xb, wxb, bx, bh, xg);

    lstm_recurrent_kernel<<<NWG, 256, 0, stream>>>(whb, xg, hbuf, out, bar);
}
